// CrossAttention_19610820674080
// MI455X (gfx1250) — matrix operand order repaired and checked by reading, not compiled, not run on hardware
//
#include <hip/hip_runtime.h>
#include <hip/hip_bf16.h>
#include <math.h>

// ---------------------------------------------------------------------------
// CDNA5 (gfx1250) fused cross-attention:
//   0a) cvt_weights : Wq/Wkv/Wproj f32 -> bf16 workspace
//   0b) cvt_kv      : kv f32 -> bf16 workspace, rows padded 512->520 so the
//                     TDM linear tile lands bank-conflict-free in LDS
//   1) qproj        : qh = (q @ Wq^T) * scale          (WMMA bf16, f32 acc)
//   2) attn_tdm     : flash attention per (b,h); kv tiles streamed into LDS
//                     with TENSOR_LOAD_TO_LDS (TDM), double buffered;
//                     K/V projected on the fly with WMMA
//      attn_fb      : fallback (ws too small): f32 kv converted in-loop
//   3) oproj        : out = attn @ Wproj^T + bproj     (WMMA bf16, f32 acc)
// ---------------------------------------------------------------------------

typedef __bf16 bf16_t;
typedef __attribute__((ext_vector_type(16))) __bf16 v16bf;
typedef __attribute__((ext_vector_type(8)))  __bf16 v8bf;
typedef __attribute__((ext_vector_type(4)))  __bf16 v4bf;
typedef __attribute__((ext_vector_type(8)))  float  v8f;
typedef __attribute__((ext_vector_type(4)))  unsigned uint32x4;
typedef __attribute__((ext_vector_type(8)))  unsigned uint32x8;

#define CA_B    32
#define CA_QN   128
#define CA_N    4096
#define CA_DIM  512
#define CA_H    8
#define CA_HD   64
#define CA_PAD  520              // padded row length for bf16 kv
#define CA_SCALE 0.125f          // 64^-0.5

// native f32 -> bf16 (RNE) -> v_cvt_pk_bf16_f32
static __device__ __forceinline__ bf16_t f2bf(float f) { return (bf16_t)f; }

// generic pointer to a __shared__ object: low 32 bits ARE the LDS byte offset
// (ISA 10.2: LDS aperture)
static __device__ __forceinline__ unsigned to_lds(const void* p) {
    return (unsigned)(size_t)p;
}

// ---- Tensor Data Mover: one instruction moves a 32x520 bf16 tile ----------
// D# group0: [1:0]=count=1 | [63:32]=lds_addr | [120:64]=global_addr | [127:126]=type=2
// D# group1: data_size=2B, tensor_dim0=520, tensor_dim1=32, tile_dim0=520,
//            tile_dim1=32, tensor_dim0_stride=520 (elements); 2D -> groups 2/3 NULL
static __device__ __forceinline__ void tdm_load_tile(unsigned ldsOff,
                                                     const bf16_t* gsrc) {
    unsigned long long ga = (unsigned long long)(size_t)gsrc;
    uint32x4 g0;
    g0[0] = 1u;                                   // count=1
    g0[1] = ldsOff;                               // lds_addr
    g0[2] = (unsigned)(ga & 0xFFFFFFFFu);         // global_addr[31:0]
    g0[3] = (unsigned)((ga >> 32) & 0x01FFFFFFu)  // global_addr[56:32]
            | (2u << 30);                         // type=2 ("image")
    uint32x8 g1;
    g1[0] = 0x00010000u;          // data_size=1 (2 bytes); mask=0; no pad/iter
    g1[1] = (unsigned)(CA_PAD) << 16;   // tensor_dim0[15:0] @ bits 63:48
    g1[2] = 32u << 16;                  // tensor_dim0 hi=0 | tensor_dim1=32
    g1[3] = (unsigned)(CA_PAD) << 16;   // tensor_dim1 hi=0 | tile_dim0=520
    g1[4] = 32u;                        // tile_dim1=32, tile_dim2=0
    g1[5] = (unsigned)(CA_PAD);         // tensor_dim0_stride[31:0]
    g1[6] = 0u;                         // stride hi | tensor_dim1_stride lo
    g1[7] = 0u;
    asm volatile("tensor_load_to_lds %0, %1"
                 :: "s"(g0), "s"(g1)
                 : "memory");
}

// A-fragment (16x32 bf16, wave32): lane row = lane&15;
// K(e) = e + (lane>=16 ? 16 : 0)  -> 16 contiguous elements.
static __device__ __forceinline__ v16bf ldA(const bf16_t* rowPtr, int c0, int lane) {
    const int kb = (lane & 16) ? 16 : 0;
    v8bf lo = *(const v8bf*)(rowPtr + c0 + kb);
    v8bf hi = *(const v8bf*)(rowPtr + c0 + kb + 8);
    v16bf a;
#pragma unroll
    for (int i = 0; i < 8; ++i) { a[i] = lo[i]; a[i + 8] = hi[i]; }
    return a;
}

// A-fragment built from f32 source (converted to bf16 on the fly)
static __device__ __forceinline__ v16bf ldA_f32(const float* rowPtr, int c0, int lane) {
    const int kb = (lane & 16) ? 16 : 0;
    float4 f0 = *(const float4*)(rowPtr + c0 + kb);
    float4 f1 = *(const float4*)(rowPtr + c0 + kb + 4);
    float4 f2 = *(const float4*)(rowPtr + c0 + kb + 8);
    float4 f3 = *(const float4*)(rowPtr + c0 + kb + 12);
    v16bf a;
    a[0]  = f2bf(f0.x); a[1]  = f2bf(f0.y); a[2]  = f2bf(f0.z); a[3]  = f2bf(f0.w);
    a[4]  = f2bf(f1.x); a[5]  = f2bf(f1.y); a[6]  = f2bf(f1.z); a[7]  = f2bf(f1.w);
    a[8]  = f2bf(f2.x); a[9]  = f2bf(f2.y); a[10] = f2bf(f2.z); a[11] = f2bf(f2.w);
    a[12] = f2bf(f3.x); a[13] = f2bf(f3.y); a[14] = f2bf(f3.z); a[15] = f2bf(f3.w);
    return a;
}

// B-fragment (32x16 bf16): lane col = lane&15; K(e) = e + (lane>=16 ? 16 : 0);
// the 16 K values are contiguous at the given pointer (16B alignment).
static __device__ __forceinline__ v16bf ldB16(const bf16_t* p) {
    v8bf lo = *(const v8bf*)p;
    v8bf hi = *(const v8bf*)(p + 8);
    v16bf b;
#pragma unroll
    for (int i = 0; i < 8; ++i) { b[i] = lo[i]; b[i + 8] = hi[i]; }
    return b;
}

static __device__ __forceinline__ v8f wmma_bf16(v16bf a, v16bf b, v8f c) {
    return __builtin_amdgcn_wmma_f32_16x16x32_bf16(
        false, a, false, b, (short)0, c, false, false);
}

// ---------------------------------------------------------------------------
// Kernel 0a: weights f32 -> bf16
// ---------------------------------------------------------------------------
__global__ void cvt_weights(const float* __restrict__ wq,
                            const float* __restrict__ wkv,
                            const float* __restrict__ wp,
                            bf16_t* __restrict__ dst) {
    const int NW = CA_DIM * CA_DIM;
    const int NKV = 2 * CA_DIM * CA_DIM;
    const int total = NW + NKV + NW;
    for (int i = blockIdx.x * blockDim.x + threadIdx.x; i < total;
         i += gridDim.x * blockDim.x) {
        float v = (i < NW) ? wq[i]
                 : (i < NW + NKV) ? wkv[i - NW]
                 : wp[i - NW - NKV];
        dst[i] = f2bf(v);
    }
}

// ---------------------------------------------------------------------------
// Kernel 0b: kv f32 [rows][512] -> bf16 [rows][520] (pad zeroed)
// ---------------------------------------------------------------------------
__global__ __launch_bounds__(128) void cvt_kv(const float* __restrict__ in,
                                              bf16_t* __restrict__ out) {
    const int row = blockIdx.x;
    const int t = threadIdx.x;
    float4 f = *(const float4*)(in + (size_t)row * CA_DIM + t * 4);
    v4bf o;
    o[0] = f2bf(f.x); o[1] = f2bf(f.y); o[2] = f2bf(f.z); o[3] = f2bf(f.w);
    *(v4bf*)(out + (size_t)row * CA_PAD + t * 4) = o;
    if (t < 2) {
        v4bf z = {};
        *(v4bf*)(out + (size_t)row * CA_PAD + CA_DIM + t * 4) = z;
    }
}

// ---------------------------------------------------------------------------
// Kernel 1: qh = (q @ Wq^T) * scale  -> bf16   [4096 x 512]
// ---------------------------------------------------------------------------
__global__ __launch_bounds__(256) void qproj_kernel(const float* __restrict__ q,
                                                    const bf16_t* __restrict__ wq_bf,
                                                    bf16_t* __restrict__ qh) {
    const int tid = threadIdx.x;
    const int wave = tid >> 5, lane = tid & 31, ln = lane & 15;
    const int half8 = (lane & 16) ? 8 : 0;
    const int koff  = (lane & 16) ? 16 : 0;
    const int Mbase = blockIdx.x * 128, Nbase = blockIdx.y * 64;

    const float*  arow = q + (size_t)(Mbase + wave * 16 + ln) * CA_DIM;
    const bf16_t* b0   = wq_bf + (size_t)(Nbase + ln) * CA_DIM;

    v8f acc0 = {}, acc1 = {}, acc2 = {}, acc3 = {};
#pragma unroll 4
    for (int c0 = 0; c0 < CA_DIM; c0 += 32) {
        v16bf A  = ldA_f32(arow, c0, lane);
        v16bf B0 = ldB16(b0 + c0 + koff);
        v16bf B1 = ldB16(b0 + (size_t)16 * CA_DIM + c0 + koff);
        v16bf B2 = ldB16(b0 + (size_t)32 * CA_DIM + c0 + koff);
        v16bf B3 = ldB16(b0 + (size_t)48 * CA_DIM + c0 + koff);
        acc0 = wmma_bf16(A, B0, acc0);
        acc1 = wmma_bf16(A, B1, acc1);
        acc2 = wmma_bf16(A, B2, acc2);
        acc3 = wmma_bf16(A, B3, acc3);
    }
#pragma unroll
    for (int v = 0; v < 8; ++v) {
        size_t r = (size_t)(Mbase + wave * 16 + v + half8) * CA_DIM + Nbase;
        qh[r + ln]      = f2bf(acc0[v] * CA_SCALE);
        qh[r + 16 + ln] = f2bf(acc1[v] * CA_SCALE);
        qh[r + 32 + ln] = f2bf(acc2[v] * CA_SCALE);
        qh[r + 48 + ln] = f2bf(acc3[v] * CA_SCALE);
    }
}

// ---------------------------------------------------------------------------
// Kernel 2 (primary): flash attention, bf16 kv tiles moved by the TDM,
// double-buffered LDS.  grid (B=32, H=8), 256 threads = 8 wave32.
// ---------------------------------------------------------------------------
__global__ __launch_bounds__(256) void attn_tdm(const bf16_t* __restrict__ kvb,
                                                const float* __restrict__ mask,
                                                const bf16_t* __restrict__ qh,
                                                const bf16_t* __restrict__ wkv_bf,
                                                bf16_t* __restrict__ attn_out) {
    __shared__ bf16_t kvTile[2][32][CA_PAD];   // 2 x 33.3 KB TDM destination
    __shared__ bf16_t kTile[32][72];           // K tile [key][d]
    __shared__ bf16_t vTileT[64][40];          // V tile transposed [d][key]
    __shared__ bf16_t pStage[8][16][32];       // per-wave P re-layout staging

    const int b = blockIdx.x, h = blockIdx.y;
    const int tid = threadIdx.x;
    const int wave = tid >> 5, lane = tid & 31, ln = lane & 15;
    const int half8 = (lane & 16) ? 8 : 0;
    const int koff  = (lane & 16) ? 16 : 0;
    const int qBase = wave * 16;

    // query fragments (16 rows x 64 head dims), loaded once
    const bf16_t* qrow = qh + (size_t)(b * CA_QN + qBase + ln) * CA_DIM + h * CA_HD;
    v16bf qA0 = ldA(qrow, 0, lane);
    v16bf qA1 = ldA(qrow, 32, lane);

    // K/V projection split: waves 0-3 -> K, 4-7 -> V; each wave: one 16-key
    // strip x two 16-dim tiles.
    const int isV = wave >> 2;
    const int ww = wave & 3;
    const int keyStrip = ww >> 1;
    const int dT0 = (ww & 1) * 2;
    const bf16_t* Wr0 = wkv_bf +
        (size_t)(isV * CA_DIM + h * CA_HD + dT0 * 16 + ln) * CA_DIM;
    const bf16_t* Wr1 = Wr0 + (size_t)16 * CA_DIM;

    const unsigned ldsBuf0 = to_lds(&kvTile[0][0][0]);
    const unsigned ldsBuf1 = to_lds(&kvTile[1][0][0]);
    const bf16_t* kvBatch = kvb + (size_t)b * CA_N * CA_PAD;

    v8f O0 = {}, O1 = {}, O2 = {}, O3 = {};
    float mstate[8], lstate[8];
#pragma unroll
    for (int v = 0; v < 8; ++v) { mstate[v] = -3.0e38f; lstate[v] = 0.0f; }

    // prologue: TDM tile 0 into buffer 0 (one instruction, wave 0 only —
    // TDM ignores EXEC, so every wave issuing would duplicate the DMA)
    if (wave == 0) tdm_load_tile(ldsBuf0, kvBatch);

#pragma unroll 1
    for (int it = 0; it < CA_N / 32; ++it) {
        const int cur = it & 1;
        if (wave == 0) __builtin_amdgcn_s_wait_tensorcnt(0);
        __syncthreads();   // publish buf[cur] to all waves; prior readers done

        // kick off TDM for next tile into the other buffer
        if (wave == 0 && it + 1 < CA_N / 32)
            tdm_load_tile(cur ? ldsBuf0 : ldsBuf1,
                          kvBatch + (size_t)(it + 1) * 32 * CA_PAD);

        // project K/V tiles: out[key][d] = sum_c kv[key][c] * Wkv[d][c]
        {
            const bf16_t* kvRow = &kvTile[cur][keyStrip * 16 + ln][0];
            v8f a0 = {}, a1 = {};
#pragma unroll 4
            for (int c0 = 0; c0 < CA_DIM; c0 += 32) {
                v16bf A  = ldA(kvRow, c0, lane);
                v16bf B0 = ldB16(Wr0 + c0 + koff);
                v16bf B1 = ldB16(Wr1 + c0 + koff);
                a0 = wmma_bf16(A, B0, a0);
                a1 = wmma_bf16(A, B1, a1);
            }
#pragma unroll
            for (int v = 0; v < 8; ++v) {
                int key = keyStrip * 16 + v + half8;
                if (!isV) {
                    kTile[key][dT0 * 16 + ln]      = f2bf(a0[v]);
                    kTile[key][dT0 * 16 + 16 + ln] = f2bf(a1[v]);
                } else {
                    vTileT[dT0 * 16 + ln][key]      = f2bf(a0[v]);
                    vTileT[dT0 * 16 + 16 + ln][key] = f2bf(a1[v]);
                }
            }
        }
        __syncthreads();

        // S = qh . K^T  (16 queries x 32 keys)
        const int n0 = it * 32;
        v16bf kb0 = ldB16(&kTile[ln][koff]);
        v16bf kb1 = ldB16(&kTile[ln][32 + koff]);
        v16bf kb2 = ldB16(&kTile[16 + ln][koff]);
        v16bf kb3 = ldB16(&kTile[16 + ln][32 + koff]);
        v8f s0 = {}, s1 = {};
        s0 = wmma_bf16(qA0, kb0, s0);
        s0 = wmma_bf16(qA1, kb1, s0);
        s1 = wmma_bf16(qA0, kb2, s1);
        s1 = wmma_bf16(qA1, kb3, s1);

        // additive mask + online softmax (shuffle reductions over 16-lane half)
        const float* mbase = mask + (size_t)(b * CA_QN + qBase) * CA_N + n0;
        float sc[8];
#pragma unroll
        for (int v = 0; v < 8; ++v) {
            const float* mr = mbase + (size_t)(v + half8) * CA_N;
            float x0 = s0[v] + mr[ln];
            float x1 = s1[v] + mr[16 + ln];
            float m = fmaxf(x0, x1);
            m = fmaxf(m, __shfl_xor(m, 8, 32));
            m = fmaxf(m, __shfl_xor(m, 4, 32));
            m = fmaxf(m, __shfl_xor(m, 2, 32));
            m = fmaxf(m, __shfl_xor(m, 1, 32));
            float mN = fmaxf(mstate[v], m);
            float scale = __expf(mstate[v] - mN);
            mstate[v] = mN;
            float e0 = __expf(x0 - mN);
            float e1 = __expf(x1 - mN);
            float rs = e0 + e1;
            rs += __shfl_xor(rs, 8, 32);
            rs += __shfl_xor(rs, 4, 32);
            rs += __shfl_xor(rs, 2, 32);
            rs += __shfl_xor(rs, 1, 32);
            lstate[v] = lstate[v] * scale + rs;
            sc[v] = scale;
            pStage[wave][v + half8][ln]      = f2bf(e0);
            pStage[wave][v + half8][16 + ln] = f2bf(e1);
        }

        // O = O*scale + P.V
        v16bf pA  = ldA(&pStage[wave][ln][0], 0, lane);
        v16bf vb0 = ldB16(&vTileT[ln][koff]);
        v16bf vb1 = ldB16(&vTileT[16 + ln][koff]);
        v16bf vb2 = ldB16(&vTileT[32 + ln][koff]);
        v16bf vb3 = ldB16(&vTileT[48 + ln][koff]);
#pragma unroll
        for (int v = 0; v < 8; ++v) {
            O0[v] *= sc[v]; O1[v] *= sc[v]; O2[v] *= sc[v]; O3[v] *= sc[v];
        }
        O0 = wmma_bf16(pA, vb0, O0);
        O1 = wmma_bf16(pA, vb1, O1);
        O2 = wmma_bf16(pA, vb2, O2);
        O3 = wmma_bf16(pA, vb3, O3);
    }

#pragma unroll
    for (int v = 0; v < 8; ++v) {
        float inv = 1.0f / lstate[v];
        size_t r = (size_t)(b * CA_QN + qBase + v + half8) * CA_DIM + h * CA_HD;
        attn_out[r + ln]      = f2bf(O0[v] * inv);
        attn_out[r + 16 + ln] = f2bf(O1[v] * inv);
        attn_out[r + 32 + ln] = f2bf(O2[v] * inv);
        attn_out[r + 48 + ln] = f2bf(O3[v] * inv);
    }
}

// ---------------------------------------------------------------------------
// Kernel 2 (fallback, small workspace): f32 kv converted in-loop
// ---------------------------------------------------------------------------
__global__ __launch_bounds__(256) void attn_fb(const float* __restrict__ kv,
                                               const float* __restrict__ mask,
                                               const bf16_t* __restrict__ qh,
                                               const bf16_t* __restrict__ wkv_bf,
                                               bf16_t* __restrict__ attn_out) {
    __shared__ bf16_t kvTile[32][520];
    __shared__ bf16_t kTile[32][72];
    __shared__ bf16_t vTileT[64][40];
    __shared__ bf16_t pStage[8][16][32];

    const int b = blockIdx.x, h = blockIdx.y;
    const int tid = threadIdx.x;
    const int wave = tid >> 5, lane = tid & 31, ln = lane & 15;
    const int half8 = (lane & 16) ? 8 : 0;
    const int koff  = (lane & 16) ? 16 : 0;
    const int qBase = wave * 16;

    const bf16_t* qrow = qh + (size_t)(b * CA_QN + qBase + ln) * CA_DIM + h * CA_HD;
    v16bf qA0 = ldA(qrow, 0, lane);
    v16bf qA1 = ldA(qrow, 32, lane);

    const int isV = wave >> 2;
    const int ww = wave & 3;
    const int keyStrip = ww >> 1;
    const int dT0 = (ww & 1) * 2;
    const bf16_t* Wr0 = wkv_bf +
        (size_t)(isV * CA_DIM + h * CA_HD + dT0 * 16 + ln) * CA_DIM;
    const bf16_t* Wr1 = Wr0 + (size_t)16 * CA_DIM;

    v8f O0 = {}, O1 = {}, O2 = {}, O3 = {};
    float mstate[8], lstate[8];
#pragma unroll
    for (int v = 0; v < 8; ++v) { mstate[v] = -3.0e38f; lstate[v] = 0.0f; }

#pragma unroll 1
    for (int n0 = 0; n0 < CA_N; n0 += 32) {
        __syncthreads();
        for (int i = tid; i < 32 * 128; i += 256) {
            int key = i >> 7, c4 = (i & 127) << 2;
            float4 f = *(const float4*)(kv +
                ((size_t)b * CA_N + n0 + key) * CA_DIM + c4);
            v4bf o;
            o[0] = f2bf(f.x); o[1] = f2bf(f.y);
            o[2] = f2bf(f.z); o[3] = f2bf(f.w);
            *(v4bf*)(&kvTile[key][c4]) = o;
        }
        __syncthreads();

        {
            const bf16_t* kvRow = &kvTile[keyStrip * 16 + ln][0];
            v8f a0 = {}, a1 = {};
#pragma unroll 4
            for (int c0 = 0; c0 < CA_DIM; c0 += 32) {
                v16bf A  = ldA(kvRow, c0, lane);
                v16bf B0 = ldB16(Wr0 + c0 + koff);
                v16bf B1 = ldB16(Wr1 + c0 + koff);
                a0 = wmma_bf16(A, B0, a0);
                a1 = wmma_bf16(A, B1, a1);
            }
#pragma unroll
            for (int v = 0; v < 8; ++v) {
                int key = keyStrip * 16 + v + half8;
                if (!isV) {
                    kTile[key][dT0 * 16 + ln]      = f2bf(a0[v]);
                    kTile[key][dT0 * 16 + 16 + ln] = f2bf(a1[v]);
                } else {
                    vTileT[dT0 * 16 + ln][key]      = f2bf(a0[v]);
                    vTileT[dT0 * 16 + 16 + ln][key] = f2bf(a1[v]);
                }
            }
        }
        __syncthreads();

        v16bf kb0 = ldB16(&kTile[ln][koff]);
        v16bf kb1 = ldB16(&kTile[ln][32 + koff]);
        v16bf kb2 = ldB16(&kTile[16 + ln][koff]);
        v16bf kb3 = ldB16(&kTile[16 + ln][32 + koff]);
        v8f s0 = {}, s1 = {};
        s0 = wmma_bf16(qA0, kb0, s0);
        s0 = wmma_bf16(qA1, kb1, s0);
        s1 = wmma_bf16(qA0, kb2, s1);
        s1 = wmma_bf16(qA1, kb3, s1);

        const float* mbase = mask + (size_t)(b * CA_QN + qBase) * CA_N + n0;
        float sc[8];
#pragma unroll
        for (int v = 0; v < 8; ++v) {
            const float* mr = mbase + (size_t)(v + half8) * CA_N;
            float x0 = s0[v] + mr[ln];
            float x1 = s1[v] + mr[16 + ln];
            float m = fmaxf(x0, x1);
            m = fmaxf(m, __shfl_xor(m, 8, 32));
            m = fmaxf(m, __shfl_xor(m, 4, 32));
            m = fmaxf(m, __shfl_xor(m, 2, 32));
            m = fmaxf(m, __shfl_xor(m, 1, 32));
            float mN = fmaxf(mstate[v], m);
            float scale = __expf(mstate[v] - mN);
            mstate[v] = mN;
            float e0 = __expf(x0 - mN);
            float e1 = __expf(x1 - mN);
            float rs = e0 + e1;
            rs += __shfl_xor(rs, 8, 32);
            rs += __shfl_xor(rs, 4, 32);
            rs += __shfl_xor(rs, 2, 32);
            rs += __shfl_xor(rs, 1, 32);
            lstate[v] = lstate[v] * scale + rs;
            sc[v] = scale;
            pStage[wave][v + half8][ln]      = f2bf(e0);
            pStage[wave][v + half8][16 + ln] = f2bf(e1);
        }

        v16bf pA  = ldA(&pStage[wave][ln][0], 0, lane);
        v16bf vb0 = ldB16(&vTileT[ln][koff]);
        v16bf vb1 = ldB16(&vTileT[16 + ln][koff]);
        v16bf vb2 = ldB16(&vTileT[32 + ln][koff]);
        v16bf vb3 = ldB16(&vTileT[48 + ln][koff]);
#pragma unroll
        for (int v = 0; v < 8; ++v) {
            O0[v] *= sc[v]; O1[v] *= sc[v]; O2[v] *= sc[v]; O3[v] *= sc[v];
        }
        O0 = wmma_bf16(pA, vb0, O0);
        O1 = wmma_bf16(pA, vb1, O1);
        O2 = wmma_bf16(pA, vb2, O2);
        O3 = wmma_bf16(pA, vb3, O3);
    }

#pragma unroll
    for (int v = 0; v < 8; ++v) {
        float inv = 1.0f / lstate[v];
        size_t r = (size_t)(b * CA_QN + qBase + v + half8) * CA_DIM + h * CA_HD;
        attn_out[r + ln]      = f2bf(O0[v] * inv);
        attn_out[r + 16 + ln] = f2bf(O1[v] * inv);
        attn_out[r + 32 + ln] = f2bf(O2[v] * inv);
        attn_out[r + 48 + ln] = f2bf(O3[v] * inv);
    }
}

// ---------------------------------------------------------------------------
// Kernel 3: out = attn @ Wproj^T + bproj  -> f32   [4096 x 512]
// ---------------------------------------------------------------------------
__global__ __launch_bounds__(256) void oproj_kernel(const bf16_t* __restrict__ x,
                                                    const bf16_t* __restrict__ wp_bf,
                                                    const float* __restrict__ bias,
                                                    float* __restrict__ out) {
    const int tid = threadIdx.x;
    const int wave = tid >> 5, lane = tid & 31, ln = lane & 15;
    const int half8 = (lane & 16) ? 8 : 0;
    const int koff  = (lane & 16) ? 16 : 0;
    const int Mbase = blockIdx.x * 128, Nbase = blockIdx.y * 64;

    const bf16_t* arow = x + (size_t)(Mbase + wave * 16 + ln) * CA_DIM;
    const bf16_t* b0   = wp_bf + (size_t)(Nbase + ln) * CA_DIM;

    float bv0 = bias[Nbase + ln];
    float bv1 = bias[Nbase + 16 + ln];
    float bv2 = bias[Nbase + 32 + ln];
    float bv3 = bias[Nbase + 48 + ln];

    v8f acc0 = {}, acc1 = {}, acc2 = {}, acc3 = {};
#pragma unroll 4
    for (int c0 = 0; c0 < CA_DIM; c0 += 32) {
        v16bf A  = ldA(arow, c0, lane);
        v16bf B0 = ldB16(b0 + c0 + koff);
        v16bf B1 = ldB16(b0 + (size_t)16 * CA_DIM + c0 + koff);
        v16bf B2 = ldB16(b0 + (size_t)32 * CA_DIM + c0 + koff);
        v16bf B3 = ldB16(b0 + (size_t)48 * CA_DIM + c0 + koff);
        acc0 = wmma_bf16(A, B0, acc0);
        acc1 = wmma_bf16(A, B1, acc1);
        acc2 = wmma_bf16(A, B2, acc2);
        acc3 = wmma_bf16(A, B3, acc3);
    }
#pragma unroll
    for (int v = 0; v < 8; ++v) {
        size_t r = (size_t)(Mbase + wave * 16 + v + half8) * CA_DIM + Nbase;
        out[r + ln]      = acc0[v] + bv0;
        out[r + 16 + ln] = acc1[v] + bv1;
        out[r + 32 + ln] = acc2[v] + bv2;
        out[r + 48 + ln] = acc3[v] + bv3;
    }
}

// ---------------------------------------------------------------------------
extern "C" void kernel_launch(void* const* d_in, const int* in_sizes, int n_in,
                              void* d_out, int out_size, void* d_ws, size_t ws_size,
                              hipStream_t stream) {
    const float* q     = (const float*)d_in[0];
    const float* kv    = (const float*)d_in[1];
    const float* mask  = (const float*)d_in[2];
    const float* Wq    = (const float*)d_in[3];
    const float* Wkv   = (const float*)d_in[4];
    const float* Wproj = (const float*)d_in[5];
    const float* bproj = (const float*)d_in[6];
    float* out = (float*)d_out;

    // workspace layout (bf16 elements):
    //   [Wq | Wkv | Wproj | qh | attn | kv_bf16(padded 520)]
    bf16_t* wbf     = (bf16_t*)d_ws;
    bf16_t* wq_bf   = wbf;                            //   262,144
    bf16_t* wkv_bf  = wbf + 262144;                   //   524,288
    bf16_t* wp_bf   = wbf + 262144 + 524288;          //   262,144
    bf16_t* qh_ws   = wbf + 1048576;                  // 2,097,152
    bf16_t* attn_ws = wbf + 1048576 + 2097152;        // 2,097,152
    bf16_t* kv_bf   = wbf + 5242880;                  // 32*4096*520 = 68,157,440
    const size_t wsNeedBig = (size_t)(5242880 + 68157440) * sizeof(bf16_t);

    cvt_weights<<<512, 256, 0, stream>>>(Wq, Wkv, Wproj, wbf);
    qproj_kernel<<<dim3(32, 8), 256, 0, stream>>>(q, wq_bf, qh_ws);

    if (ws_size >= wsNeedBig) {
        cvt_kv<<<CA_B * CA_N, 128, 0, stream>>>(kv, kv_bf);
        attn_tdm<<<dim3(32, 8), 256, 0, stream>>>(kv_bf, mask, qh_ws, wkv_bf, attn_ws);
    } else {
        attn_fb<<<dim3(32, 8), 256, 0, stream>>>(kv, mask, qh_ws, wkv_bf, attn_ws);
    }

    oproj_kernel<<<dim3(32, 8), 256, 0, stream>>>(attn_ws, wp_bf, bproj, out);

    (void)in_sizes; (void)n_in; (void)out_size; (void)ws_size;
}
